// ProtoNet_8675833938767
// MI455X (gfx1250) — matrix operand order repaired and checked by reading, not compiled, not run on hardware
//
#include <hip/hip_runtime.h>
#include <hip/hip_bf16.h>

// ---------------------------------------------------------------------------
// ProtoNet forward on MI455X (gfx1250, wave32):
//   prot = class-mean(xs)            (scatter + normalize)
//   out  = -sqrt(max(|q|^2 + |p|^2 - 2 q.p, 0))
// Cross term via bf16 WMMA with hi/lo fp32-split (hi*hi + hi*lo + lo*hi).
// GEMM mainloop double-buffers LDS; operand staging uses CDNA5 async
// global->LDS copies (ASYNCcnt) when the builtin is available.
// ---------------------------------------------------------------------------

#define NCLS 4096
#define QN   4096
#define DDIM 512
#define SHOTS_TOTAL (NCLS * 4)

#define BM 128
#define BN 128
#define BK 32
#define KSTEPS (DDIM / BK)   // 16

typedef __attribute__((ext_vector_type(16))) __bf16 v16bf;
typedef __attribute__((ext_vector_type(8)))  __bf16 v8bf;
typedef __attribute__((ext_vector_type(8)))  float  v8f;
typedef int v4i __attribute__((vector_size(16)));   // 128-bit async-copy payload

#if defined(__has_builtin)
#if __has_builtin(__builtin_amdgcn_global_load_async_to_lds_b128) && \
    __has_builtin(__builtin_amdgcn_s_wait_asynccnt)
#define USE_ASYNC_LDS 1
#endif
#endif
#ifndef USE_ASYNC_LDS
#define USE_ASYNC_LDS 0
#endif

__device__ __forceinline__ unsigned short f32_to_bf16_rne(float f) {
    unsigned u = __builtin_bit_cast(unsigned, f);
    unsigned r = u + 0x7FFFu + ((u >> 16) & 1u);
    return (unsigned short)(r >> 16);
}
__device__ __forceinline__ float bf16_to_f32(unsigned short h) {
    return __builtin_bit_cast(float, (unsigned)h << 16);
}

__device__ __forceinline__ void lds_copy16(unsigned short* dst_lds,
                                           const unsigned short* src_glob) {
#if USE_ASYNC_LDS
    __builtin_amdgcn_global_load_async_to_lds_b128(
        (__attribute__((address_space(1))) v4i*)src_glob,
        (__attribute__((address_space(3))) v4i*)dst_lds,
        /*offset=*/0, /*cpol=*/0);
#else
    *(uint4*)dst_lds = *(const uint4*)src_glob;
#endif
}

// ---------------------------------------------------------------------------
// 1) zero prototype sums + counts (contiguous region)
// ---------------------------------------------------------------------------
__global__ __launch_bounds__(256) void zero_f32_kernel(float* __restrict__ p, int n) {
    int i = blockIdx.x * 256 + threadIdx.x;
    if (i < n) p[i] = 0.0f;
}

// ---------------------------------------------------------------------------
// 2) scatter-add support rows into class sums (generic segment-sum)
// ---------------------------------------------------------------------------
__global__ __launch_bounds__(256) void scatter_kernel(const float* __restrict__ xs,
                                                      const int*   __restrict__ ys,
                                                      float* __restrict__ sums,
                                                      float* __restrict__ counts) {
    int s = blockIdx.x;
    int c = ys[s];
    const float* r = xs + (size_t)s * DDIM;
    float* dst = sums + (size_t)c * DDIM;
    for (int d = threadIdx.x; d < DDIM; d += 256)
        atomicAdd(&dst[d], r[d]);
    if (threadIdx.x == 0)
        atomicAdd(&counts[c], 1.0f);
}

// ---------------------------------------------------------------------------
// 3) per-row finalize: optional mean-scale, bf16 hi/lo split, squared norm
// ---------------------------------------------------------------------------
__global__ __launch_bounds__(256) void finalize_rows_kernel(const float* __restrict__ src,
                                                            const float* __restrict__ counts,
                                                            unsigned short* __restrict__ hi,
                                                            unsigned short* __restrict__ lo,
                                                            float* __restrict__ sq) {
    __shared__ float red[256];
    int row = blockIdx.x;
    float scale = 1.0f;
    if (counts) {
        float c = counts[row];
        scale = (c > 0.0f) ? (1.0f / c) : 0.0f;
    }
    const float* r = src + (size_t)row * DDIM;
    unsigned short* hrow = hi + (size_t)row * DDIM;
    unsigned short* lrow = lo + (size_t)row * DDIM;
    float ss = 0.0f;
    for (int d = threadIdx.x; d < DDIM; d += 256) {
        float x = r[d] * scale;
        ss += x * x;
        unsigned short h = f32_to_bf16_rne(x);
        float hf = bf16_to_f32(h);
        unsigned short l = f32_to_bf16_rne(x - hf);
        hrow[d] = h;
        lrow[d] = l;
    }
    red[threadIdx.x] = ss;
    __syncthreads();
    for (int s = 128; s > 0; s >>= 1) {
        if (threadIdx.x < s) red[threadIdx.x] += red[threadIdx.x + s];
        __syncthreads();
    }
    if (threadIdx.x == 0) sq[row] = red[0];
}

// ---------------------------------------------------------------------------
// 4) WMMA GEMM + distance epilogue
//    block tile 128x128, 8 wave32 waves, each wave: 32 rows x 64 cols
//    (2x4 tiles of 16x16). K staged through double-buffered LDS panels
//    (2 x 4 x 128x32 bf16 = 64 KB) filled with async global->LDS copies.
// ---------------------------------------------------------------------------
__global__ __launch_bounds__(256) void gemm_dist_kernel(const unsigned short* __restrict__ a_hi,
                                                        const unsigned short* __restrict__ a_lo,
                                                        const unsigned short* __restrict__ b_hi,
                                                        const unsigned short* __restrict__ b_lo,
                                                        const float* __restrict__ q_sq,
                                                        const float* __restrict__ p_sq,
                                                        float* __restrict__ out) {
    // [buffer][panel][elem]; panel 0=A_hi, 1=A_lo, 2=B_hi, 3=B_lo
    __shared__ __attribute__((aligned(16))) unsigned short smem[2][4][BM * BK];

    const int tid    = threadIdx.x;
    const int lane   = tid & 31;
    const int wid    = tid >> 5;
    const int wave_m = wid & 3;   // 4 wave rows  -> 32 rows each
    const int wave_n = wid >> 2;  // 2 wave cols  -> 64 cols each
    const int lmod   = lane & 15;
    const int lhalf  = lane >> 4;

    const int rowBase = blockIdx.y * BM;
    const int colBase = blockIdx.x * BN;

    v8f zero;
#pragma unroll
    for (int i = 0; i < 8; ++i) zero[i] = 0.0f;

    v8f acc[2][4];
#pragma unroll
    for (int mi = 0; mi < 2; ++mi)
#pragma unroll
        for (int ni = 0; ni < 4; ++ni) acc[mi][ni] = zero;

    // issue one K-panel's copies (8 x 16B per thread -> 32 KB per buffer)
    auto issue_tile = [&](int kt, int buf) {
#pragma unroll
        for (int pass = 0; pass < 2; ++pass) {
            int ch = tid + pass * 256;        // 0..511
            int r  = ch >> 2;                 // row in tile 0..127
            int cc = (ch & 3) * 8;            // k-offset 0/8/16/24
            size_t ga = (size_t)(rowBase + r) * DDIM + kt * BK + cc;
            size_t gb = (size_t)(colBase + r) * DDIM + kt * BK + cc;
            lds_copy16(&smem[buf][0][r * BK + cc], &a_hi[ga]);
            lds_copy16(&smem[buf][1][r * BK + cc], &a_lo[ga]);
            lds_copy16(&smem[buf][2][r * BK + cc], &b_hi[gb]);
            lds_copy16(&smem[buf][3][r * BK + cc], &b_lo[gb]);
        }
    };

    // consume one K-panel: build fragments per CDNA5 bf16 layouts and WMMA
    auto compute_tile = [&](int cur) {
        // A (16x32 bf16): lanes 0-15 K[0..7]+K[16..23], lanes 16-31 K[8..15]+K[24..31]
        v16bf afrag[2][2];
#pragma unroll
        for (int mi = 0; mi < 2; ++mi) {
            int row = wave_m * 32 + mi * 16 + lmod;
            int c0  = lhalf * 8;
#pragma unroll
            for (int hl = 0; hl < 2; ++hl) {
                const unsigned short* base = &smem[cur][hl][row * BK];
                v8bf p0 = *(const v8bf*)&base[c0];
                v8bf p1 = *(const v8bf*)&base[16 + c0];
                afrag[mi][hl] = __builtin_shufflevector(p0, p1, 0, 1, 2, 3, 4, 5, 6, 7,
                                                        8, 9, 10, 11, 12, 13, 14, 15);
            }
        }
        // B (32x16 bf16): lanes 0-15 K[0..7]+K[16..23], lanes 16-31 K[8..15]+K[24..31]
        v16bf bfrag[4][2];
#pragma unroll
        for (int ni = 0; ni < 4; ++ni) {
            int rowc = wave_n * 64 + ni * 16 + lmod;
            int kh   = lhalf * 8;
#pragma unroll
            for (int hl = 0; hl < 2; ++hl) {
                const unsigned short* base = &smem[cur][2 + hl][rowc * BK + kh];
                v8bf p0 = *(const v8bf*)&base[0];
                v8bf p1 = *(const v8bf*)&base[16];
                bfrag[ni][hl] = __builtin_shufflevector(p0, p1, 0, 1, 2, 3, 4, 5, 6, 7,
                                                        8, 9, 10, 11, 12, 13, 14, 15);
            }
        }
        // hi*hi + hi*lo + lo*hi  (drop lo*lo, ~2^-16 relative)
#pragma unroll
        for (int mi = 0; mi < 2; ++mi)
#pragma unroll
            for (int ni = 0; ni < 4; ++ni) {
                acc[mi][ni] = __builtin_amdgcn_wmma_f32_16x16x32_bf16(
                    false, afrag[mi][0], false, bfrag[ni][0], (short)0, acc[mi][ni], false, false);
                acc[mi][ni] = __builtin_amdgcn_wmma_f32_16x16x32_bf16(
                    false, afrag[mi][0], false, bfrag[ni][1], (short)0, acc[mi][ni], false, false);
                acc[mi][ni] = __builtin_amdgcn_wmma_f32_16x16x32_bf16(
                    false, afrag[mi][1], false, bfrag[ni][0], (short)0, acc[mi][ni], false, false);
            }
    };

    // ---- software pipeline: prefetch next panel while computing current
    issue_tile(0, 0);
    for (int kt = 0; kt < KSTEPS - 1; ++kt) {
        int cur = kt & 1;
        issue_tile(kt + 1, cur ^ 1);
#if USE_ASYNC_LDS
        __builtin_amdgcn_s_wait_asynccnt(8);   // current panel landed; next in flight
#endif
        __syncthreads();
        compute_tile(cur);
        __syncthreads();                        // readers done before buffer reuse
    }
#if USE_ASYNC_LDS
    __builtin_amdgcn_s_wait_asynccnt(0);
#endif
    __syncthreads();
    compute_tile((KSTEPS - 1) & 1);

    // ---- epilogue: C/D layout -> lane L: N = L%16, M = v + 8*(L/16)
#pragma unroll
    for (int mi = 0; mi < 2; ++mi) {
#pragma unroll
        for (int v = 0; v < 8; ++v) {
            int row = rowBase + wave_m * 32 + mi * 16 + v + lhalf * 8;
            float qs = q_sq[row];
#pragma unroll
            for (int ni = 0; ni < 4; ++ni) {
                int col = colBase + wave_n * 64 + ni * 16 + lmod;
                float ps = p_sq[col];
                float cr = acc[mi][ni][v];
                float d2 = qs + ps - 2.0f * cr;
                d2 = d2 > 0.0f ? d2 : 0.0f;
                out[(size_t)row * NCLS + col] = -__builtin_sqrtf(d2);
            }
        }
    }
}

// ---------------------------------------------------------------------------
// launch
// ---------------------------------------------------------------------------
extern "C" void kernel_launch(void* const* d_in, const int* in_sizes, int n_in,
                              void* d_out, int out_size, void* d_ws, size_t ws_size,
                              hipStream_t stream) {
    const float* xs = (const float*)d_in[0];
    const int*   ys = (const int*)d_in[1];
    const float* xq = (const float*)d_in[2];
    float* out = (float*)d_out;

    char* ws = (char*)d_ws;
    // workspace layout (all 16B aligned)
    float*          proto_sum = (float*)(ws);                                  //  8 MB
    float*          counts    = (float*)(ws + 8388608);                        // 16 KB (contiguous after sums)
    unsigned short* prot_hi   = (unsigned short*)(ws + 8388608 + 16384);       //  4 MB
    unsigned short* prot_lo   = (unsigned short*)(ws + 12599296);              //  4 MB
    unsigned short* xq_hi     = (unsigned short*)(ws + 16793600);              //  4 MB
    unsigned short* xq_lo     = (unsigned short*)(ws + 20987904);              //  4 MB
    float*          p_sq      = (float*)(ws + 25182208);                       // 16 KB
    float*          q_sq      = (float*)(ws + 25198592);                       // 16 KB

    // 1) zero sums + counts (contiguous: NCLS*DDIM + NCLS floats)
    const int nzero = NCLS * DDIM + NCLS;
    zero_f32_kernel<<<(nzero + 255) / 256, 256, 0, stream>>>(proto_sum, nzero);

    // 2) scatter support features into class sums
    scatter_kernel<<<SHOTS_TOTAL, 256, 0, stream>>>(xs, ys, proto_sum, counts);

    // 3) finalize prototypes (mean + split + |p|^2) and queries (split + |q|^2)
    finalize_rows_kernel<<<NCLS, 256, 0, stream>>>(proto_sum, counts, prot_hi, prot_lo, p_sq);
    finalize_rows_kernel<<<QN, 256, 0, stream>>>(xq, nullptr, xq_hi, xq_lo, q_sq);

    // 4) WMMA GEMM + distance epilogue
    dim3 grid(NCLS / BN, QN / BM);
    gemm_dist_kernel<<<grid, 256, 0, stream>>>(xq_hi, xq_lo, prot_hi, prot_lo,
                                               q_sq, p_sq, out);
}
